// Net_30374008717896
// MI455X (gfx1250) — matrix operand order repaired and checked by reading, not compiled, not run on hardware
//
#include <hip/hip_runtime.h>
#include <hip/hip_bf16.h>

typedef __attribute__((ext_vector_type(16))) _Float16 v16h;
typedef __attribute__((ext_vector_type(8)))  _Float16 v8h;
typedef __attribute__((ext_vector_type(8)))  float    v8f;

constexpr int Hc   = 50;        // hidden
constexpr int Gc   = 150;       // 3*H gate columns
constexpr int NT   = 10;        // N tiles of 16 (padded to 160)
constexpr int KP   = 64;        // K padded 50 -> 64 (two K=32 WMMA steps)
constexpr int GHS  = 164;       // gh LDS row stride (pad vs 160 to avoid bank conflicts)
constexpr int WAVES = 2;
constexpr int TILES = 4;
constexpr int BD    = WAVES * 32;                 // 64 threads
constexpr int ROWS_PER_BLOCK = WAVES * TILES * 16; // 128

__global__ __launch_bounds__(BD) void gru_ac_kernel(
    const float* __restrict__ s, const float* __restrict__ h,
    const float* __restrict__ w_ih, const float* __restrict__ w_hh,
    const float* __restrict__ b_ih, const float* __restrict__ b_hh,
    const float* __restrict__ actor_w, const float* __restrict__ actor_b,
    const float* __restrict__ critic_w, const float* __restrict__ critic_b,
    float* __restrict__ out, int Btot)
{
    // B operand of gh-GEMM staged in WMMA lane layout:
    // bstage[(((kstep*2 + khalf)*NT + t)*16 + n)*16 + i] = w_hh[t*16+n][kstep*32+khalf*16+i]
    __shared__ __align__(32) _Float16 bstage[2 * 2 * NT * 16 * 16]; // 10240 halves
    __shared__ float wih_s[Gc * 2];
    __shared__ float bias_s[Gc];    // b_ih + b_hh folded (gates only ever see the sum)
    __shared__ float head_s[160];   // [0..49] critic_w, [50..149] actor_w, [150..151] actor_b, [152] critic_b
    __shared__ __align__(32) _Float16 hstage[WAVES][16 * KP];
    __shared__ float gh_s[WAVES][16 * GHS];
    __shared__ float hn_s[WAVES][16 * Hc];
    __shared__ float sst[WAVES][16 * 2];

    const int tid  = threadIdx.x;
    const int wave = tid >> 5;
    const int lane = tid & 31;

    // ---------------- stage weights (once per block; hits L2, ~30KB) -------------
    for (int i = tid; i < Gc * 2; i += BD) wih_s[i] = w_ih[i];
    for (int i = tid; i < Gc;     i += BD) bias_s[i] = b_ih[i] + b_hh[i];
    for (int i = tid; i < 160; i += BD) {
        float v = 0.0f;
        if      (i <  50) v = critic_w[i];
        else if (i < 150) v = actor_w[i - 50];
        else if (i == 150) v = actor_b[0];
        else if (i == 151) v = actor_b[1];
        else if (i == 152) v = critic_b[0];
        head_s[i] = v;
    }
    for (int e = tid; e < 2 * 2 * NT * 16 * 16; e += BD) {
        int i  = e & 15;
        int n  = (e >> 4) & 15;
        int r2 = e >> 8;            // 0..39
        int t  = r2 % NT;
        int sk = r2 / NT;           // 0..3
        int kh = sk & 1, sv = sk >> 1;
        int k   = sv * 32 + kh * 16 + i;
        int col = t * 16 + n;
        float v = (k < Hc && col < Gc) ? w_hh[col * Hc + k] : 0.0f;
        bstage[e] = (_Float16)v;
    }
    __syncthreads();

    float* hOut  = out;
    float* cOut  = out + (size_t)Btot * 50;
    float* piOut = out + (size_t)Btot * 51;

    for (int tile = 0; tile < TILES; ++tile) {
        const int rbase  = (blockIdx.x * WAVES + wave) * (TILES * 16) + tile * 16;
        const bool active = (rbase + 16) <= Btot;   // wave-uniform
        __syncthreads();

        // prefetch next tile's h rows (3.2KB, 128B-strided lanes cover it)
        if (tile + 1 < TILES && (rbase + 32) <= Btot) {
            const float* pf = h + (size_t)(rbase + 16) * Hc + lane * 32;
            __builtin_prefetch(pf, 0, 1);
        }

        // ---- stage s (16 rows x 2) and h (16 rows x 50 -> f16, K padded to 64) ----
        {
            int m = lane >> 1, c = lane & 1;
            sst[wave][lane] = active ? s[(size_t)(rbase + m) * 2 + c] : 0.0f;
        }
        for (int idx = 0; idx < 32; ++idx) {
            int e = idx * 32 + lane;
            int m = e >> 6, k = e & 63;
            float v = (active && k < Hc) ? h[(size_t)(rbase + m) * Hc + k] : 0.0f;
            hstage[wave][e] = (_Float16)v;
        }
        __syncthreads();

        // ---- A operands (16x32 f16 layout): lanes 0-15 row m halves K {0..15},
        //      lanes 16-31 same row, halves K {16..31} ----
        const _Float16* hrow = &hstage[wave][(lane & 15) * KP];
        const int b0 = (lane < 16) ? 0 : 16;
        v16h a0, a1;
        a0 = *(const v16h*)(hrow + b0);
        a1 = *(const v16h*)(hrow + 32 + b0);

        const int kh = lane >> 4, nn = lane & 15;
        for (int t = 0; t < NT; ++t) {
            v8f acc = {};   // zero C -> inline-0 SRC2 form; biases folded into bias_s
            const v16h bm0 = *(const v16h*)&bstage[(((0 * 2 + kh) * NT + t) * 16 + nn) * 16];
            const v16h bm1 = *(const v16h*)&bstage[(((1 * 2 + kh) * NT + t) * 16 + nn) * 16];
            acc = __builtin_amdgcn_wmma_f32_16x16x32_f16(false, a0, false, bm0,
                                                         (short)0, acc, false, false);
            acc = __builtin_amdgcn_wmma_f32_16x16x32_f16(false, a1, false, bm1,
                                                         (short)0, acc, false, false);
            #pragma unroll
            for (int j = 0; j < 8; ++j)          // D layout: row j+8*(lane/16), col lane%16
                gh_s[wave][(j + (kh << 3)) * GHS + t * 16 + nn] = acc[j];
        }
        __syncthreads();

        // ---- gate math: 16 rows x 50 = 800 elements = 25 per lane ----
        for (int it = 0; it < 25; ++it) {
            int e = it * 32 + lane;
            int b = e / Hc, j = e - b * Hc;
            float s0v = sst[wave][b * 2], s1v = sst[wave][b * 2 + 1];
            float gir = fmaf(s0v, wih_s[2 * j],            fmaf(s1v, wih_s[2 * j + 1],            bias_s[j]));
            float giz = fmaf(s0v, wih_s[2 * (Hc + j)],     fmaf(s1v, wih_s[2 * (Hc + j) + 1],     bias_s[Hc + j]));
            float gin = fmaf(s0v, wih_s[2 * (2 * Hc + j)], fmaf(s1v, wih_s[2 * (2 * Hc + j) + 1], bias_s[2 * Hc + j]));
            const float* ghrow = &gh_s[wave][b * GHS];
            float r   = 1.0f / (1.0f + __expf(-(gir + ghrow[j])));
            float z   = 1.0f / (1.0f + __expf(-(giz + ghrow[Hc + j])));
            float pre = gin + r * ghrow[2 * Hc + j];
            float nt  = 2.0f / (1.0f + __expf(-2.0f * pre)) - 1.0f;  // tanh
            float hp  = active ? h[(size_t)(rbase + b) * Hc + j] : 0.0f;
            float hnew = fmaf(z, hp - nt, nt);                       // (1-z)*n + z*h
            if (active) hOut[(size_t)(rbase + b) * Hc + j] = hnew;
            hn_s[wave][b * Hc + j] = hnew;
        }
        __syncthreads();

        // ---- heads: 2 lanes per row, shuffle-combine, 2-way softmax ----
        {
            int row = lane >> 1, hf = lane & 1;
            const float* hr = &hn_s[wave][row * Hc];
            float pc = 0.f, pa0 = 0.f, pa1 = 0.f;
            #pragma unroll
            for (int q = 0; q < 25; ++q) {
                int j = hf * 25 + q;
                float v = hr[j];
                pc  = fmaf(v, head_s[j],       pc);
                pa0 = fmaf(v, head_s[50 + j],  pa0);
                pa1 = fmaf(v, head_s[100 + j], pa1);
            }
            pc  += __shfl_xor(pc, 1);
            pa0 += __shfl_xor(pa0, 1);
            pa1 += __shfl_xor(pa1, 1);
            if (hf == 0 && active) {
                size_t rg = (size_t)rbase + row;
                cOut[rg] = pc + head_s[152];
                float l0 = pa0 + head_s[150];
                float l1 = pa1 + head_s[151];
                float m  = fmaxf(l0, l1);
                float e0 = __expf(l0 - m), e1 = __expf(l1 - m);
                float inv = 1.0f / (e0 + e1);
                piOut[rg * 2 + 0] = e0 * inv;
                piOut[rg * 2 + 1] = e1 * inv;
            }
        }
    }
}

extern "C" void kernel_launch(void* const* d_in, const int* in_sizes, int n_in,
                              void* d_out, int out_size, void* d_ws, size_t ws_size,
                              hipStream_t stream) {
    const float* s        = (const float*)d_in[0];
    const float* h        = (const float*)d_in[1];
    const float* w_ih     = (const float*)d_in[2];
    const float* w_hh     = (const float*)d_in[3];
    const float* b_ih     = (const float*)d_in[4];
    const float* b_hh     = (const float*)d_in[5];
    const float* actor_w  = (const float*)d_in[6];
    const float* actor_b  = (const float*)d_in[7];
    const float* critic_w = (const float*)d_in[8];
    const float* critic_b = (const float*)d_in[9];
    float* out = (float*)d_out;

    int B = in_sizes[0] / 2;   // s is [B, 2]
    int grid = (B + ROWS_PER_BLOCK - 1) / ROWS_PER_BLOCK;
    gru_ac_kernel<<<grid, BD, 0, stream>>>(s, h, w_ih, w_hh, b_ih, b_hh,
                                           actor_w, actor_b, critic_w, critic_b,
                                           out, B);
}
